// MultiHeadAttention_40492951666703
// MI455X (gfx1250) — matrix operand order repaired and checked by reading, not compiled, not run on hardware
//
#include <hip/hip_runtime.h>
#include <hip/hip_bf16.h>

typedef _Float16 half_t;
typedef __attribute__((ext_vector_type(16))) _Float16 v16h;
typedef __attribute__((ext_vector_type(8)))  _Float16 v8h;
typedef __attribute__((ext_vector_type(2)))  _Float16 v2h;
typedef __attribute__((ext_vector_type(8)))  float    v8f;
typedef __attribute__((ext_vector_type(4)))  float    v4f;

#define B_  4
#define S_  2048
#define E_  1024
#define H_  16
#define DH_ 64

#define LOG2E 1.44269504088896340736f

// LDS row strides (halves), all fragment reads 16B-aligned.
#define KS_STRIDE 80    // K tile: 128 keys x 64 d (+16 pad)
#define VT_STRIDE 144   // V tile: 64 d x 128 key-slots (+16 pad)
#define PS_STRIDE 40    // P tile: 16 rows x 32 key-slots (+8 pad)
#define AS_STRIDE 1032  // proj A tile: 16 rows x 1024 k (+8 pad)

__device__ __forceinline__ float fast_tanh(float x) {
#if __has_builtin(__builtin_amdgcn_tanhf)
  return __builtin_amdgcn_tanhf(x);
#elif __has_builtin(__builtin_amdgcn_tanh_f32)
  return __builtin_amdgcn_tanh_f32(x);
#else
  return tanhf(x);
#endif
}

__device__ __forceinline__ float fast_exp2(float x) {
#if __has_builtin(__builtin_amdgcn_exp2f)
  return __builtin_amdgcn_exp2f(x);   // raw v_exp_f32 (base 2)
#else
  return exp2f(x);
#endif
}

__device__ __forceinline__ v8f wmma_f16(v16h a, v16h b, v8f c) {
  return __builtin_amdgcn_wmma_f32_16x16x32_f16(false, a, false, b, (short)0, c,
                                                false, false);
}

// 16-half fragment from two 16B-aligned 8-half loads.
__device__ __forceinline__ v16h ld16(const half_t* p) {
  v8h lo = *(const v8h*)p;
  v8h hi = *(const v8h*)(p + 8);
  v16h r;
#pragma unroll
  for (int e = 0; e < 8; ++e) { r[e] = lo[e]; r[8 + e] = hi[e]; }
  return r;
}

// ---------------------------------------------------------------------------
// Kernel 1: soft-capped flash attention, DH=64 heads.
// Soft-cap => logits bounded by +/-C: softmax uses a FIXED max of C (no
// running max / rescale), row sums deferred to one final butterfly.
// p = exp2(fma(tanh(u), C*log2e, bias)), bias = masked ? -inf : -C*log2e.
// Block = 8 waves x 16 query rows; grid = B * H * (S/128).
// ---------------------------------------------------------------------------
__global__ __launch_bounds__(256)
void attn_kernel(const float* __restrict__ q, const float* __restrict__ k,
                 const float* __restrict__ v, const unsigned char* __restrict__ mask,
                 const float* __restrict__ scale_p, const float* __restrict__ logC_p,
                 half_t* __restrict__ attn_out) {
  __shared__ __attribute__((aligned(16))) half_t Ks[128 * KS_STRIDE]; // [key][d]
  __shared__ __attribute__((aligned(16))) half_t Vt[64 * VT_STRIDE];  // [d][slot]
  __shared__ __attribute__((aligned(16))) half_t Ps[8 * 16 * PS_STRIDE];
  __shared__ float smask[128];   // additive log2-domain bias

  const int tid  = threadIdx.x;
  const int wave = tid >> 5;
  const int lane = tid & 31;
  const int hf   = lane >> 4;
  const int l16  = lane & 15;

  const int bx = blockIdx.x;
  const int qt = bx & 15;
  const int h  = (bx >> 4) & 15;
  const int b  = bx >> 8;
  const int q0 = qt * 128 + wave * 16;

  const float sc   = scale_p[0] * 0.125f;      // scale / sqrt(DH), folded into Q
  const float expC = __expf(logC_p[0]);        // soft-cap C
  const float c2   = expC * LOG2E;             // C * log2(e)

  // Q A-fragments, pre-scaled by sc. A layout: k = e + hf*16.
  v16h a_q[2];
  const float* qrow = q + ((size_t)b * S_ + q0 + l16) * E_ + h * DH_;
#pragma unroll
  for (int f = 0; f < 2; ++f) {
    const int d0 = f * 32 + hf * 16;
#pragma unroll
    for (int e = 0; e < 8; ++e) a_q[f][e]     = (half_t)(qrow[d0 + e] * sc);
#pragma unroll
    for (int e = 0; e < 8; ++e) a_q[f][8 + e] = (half_t)(qrow[d0 + 8 + e] * sc);
  }

  float run_l[8];
  v8f o[4];
#pragma unroll
  for (int r = 0; r < 8; ++r) run_l[r] = 0.f;
#pragma unroll
  for (int g = 0; g < 4; ++g) { v8f z = {}; o[g] = z; }

  for (int kc = 0; kc < S_ / 128; ++kc) {
    const int kbase = kc * 128;
    // ---- stage K (row-major) and V (d-major, k-slot-permuted) as f16 ----
#pragma unroll
    for (int it = 0; it < 8; ++it) {
      const int idx = it * 256 + tid;       // 0..2047
      const int row = idx >> 4;             // key 0..127
      const int c4  = (idx & 15) << 2;      // d 0..60
      const size_t goff = ((size_t)b * S_ + kbase + row) * E_ + h * DH_ + c4;
      const v4f kv = *(const v4f*)(k + goff);
      const v4f vv = *(const v4f*)(v + goff);
      if (kc + 1 < S_ / 128) {              // prefetch next chunk
        __builtin_prefetch(k + goff + (size_t)128 * E_, 0, 1);
        __builtin_prefetch(v + goff + (size_t)128 * E_, 0, 1);
      }
      // slot permutation within each 32-key group: loc -> 2*(loc&15)+(loc>>4)
      const int grp = row >> 5, loc = row & 31;
      const int colV = grp * 32 + ((loc & 15) * 2 + (loc >> 4));
#pragma unroll
      for (int j = 0; j < 4; ++j) {
        Ks[row * KS_STRIDE + c4 + j]    = (half_t)kv[j];
        Vt[(c4 + j) * VT_STRIDE + colV] = (half_t)vv[j];
      }
    }
    if (tid < 128)
      smask[tid] = mask[(size_t)b * S_ + kbase + tid] ? -__builtin_inff() : -c2;
    __syncthreads();

#pragma unroll
    for (int g2 = 0; g2 < 4; ++g2) {
      const int ko = g2 * 32;
      // ---- all K B-fragments + biases up front (one dscnt wait) ----
      const half_t* krow0 = &Ks[(ko + l16) * KS_STRIDE];
      const half_t* krow1 = &Ks[(ko + 16 + l16) * KS_STRIDE];
      v16h bk00 = ld16(krow0 + hf * 16);
      v16h bk01 = ld16(krow0 + 32 + hf * 16);
      v16h bk10 = ld16(krow1 + hf * 16);
      v16h bk11 = ld16(krow1 + 32 + hf * 16);
      const float bias0 = smask[ko + l16];
      const float bias1 = smask[ko + 16 + l16];

      v8f s0 = {}, s1 = {};
      s0 = wmma_f16(a_q[0], bk00, s0);
      s0 = wmma_f16(a_q[1], bk01, s0);
      s1 = wmma_f16(a_q[0], bk10, s1);
      s1 = wmma_f16(a_q[1], bk11, s1);

      // V B-fragments issued now: fills the WMMA-drain / hazard window.
      v16h b_v[4];
#pragma unroll
      for (int g = 0; g < 4; ++g)
        b_v[g] = ld16(&Vt[(g * 16 + l16) * VT_STRIDE + ko + hf * 16]);

      // ---- softmax weights: tanh -> fma(log2 domain) -> v_exp ----
      float pex0[8], pex1[8];
#pragma unroll
      for (int r = 0; r < 8; ++r)
        pex0[r] = fast_exp2(fmaf(fast_tanh(s0[r]), c2, bias0));
#pragma unroll
      for (int r = 0; r < 8; ++r)
        pex1[r] = fast_exp2(fmaf(fast_tanh(s1[r]), c2, bias1));

      // per-lane row-sum accumulation; P packed 2 halves per b32 store
      half_t* prow_w = &Ps[(wave * 16 + hf * 8) * PS_STRIDE + 2 * l16];
#pragma unroll
      for (int r = 0; r < 8; ++r) {
        run_l[r] += pex0[r] + pex1[r];
        v2h pp;
        pp[0] = (half_t)pex0[r];
        pp[1] = (half_t)pex1[r];
        *(v2h*)(prow_w + r * PS_STRIDE) = pp;
      }
      __builtin_amdgcn_wave_barrier();
      asm volatile("s_wait_dscnt 0x0" ::: "memory");

      // A-fragment of P (16 x 32 slots)
      const half_t* prow_r = &Ps[(wave * 16 + l16) * PS_STRIDE];
      v8h plo = *(const v8h*)(prow_r + hf * 16);
      v8h phi = *(const v8h*)(prow_r + hf * 16 + 8);
      v16h a_p;
#pragma unroll
      for (int e = 0; e < 8; ++e) { a_p[e] = plo[e]; a_p[8 + e] = phi[e]; }

#pragma unroll
      for (int g = 0; g < 4; ++g) o[g] = wmma_f16(a_p, b_v[g], o[g]);
    }
    __syncthreads();
  }

  // one deferred row-sum butterfly across each half-wave's 16 lanes
#pragma unroll
  for (int off = 1; off < 16; off <<= 1)
#pragma unroll
    for (int r = 0; r < 8; ++r) run_l[r] += __shfl_xor(run_l[r], off, 32);
  float inv_l[8];
#pragma unroll
  for (int r = 0; r < 8; ++r) inv_l[r] = 1.0f / run_l[r];

#pragma unroll
  for (int g = 0; g < 4; ++g)
#pragma unroll
    for (int r = 0; r < 8; ++r) {
      const int m = r + hf * 8;
      attn_out[((size_t)b * S_ + q0 + m) * E_ + h * DH_ + g * 16 + l16] =
          (half_t)(o[g][r] * inv_l[r]);
    }
}

// ---------------------------------------------------------------------------
// Kernel 2: out[i][j] = sum_e A[i][e] * W[j][e].  M=8192, N=K=1024.
// Block = 8 waves sharing one 16-row A tile staged in LDS; wave w does
// col tile nt = (bx&7)*8 + w.  Grid = 512 * 8 / 16 = 256 blocks.
// ---------------------------------------------------------------------------
__global__ __launch_bounds__(256)
void proj_kernel(const half_t* __restrict__ A, const float* __restrict__ W,
                 float* __restrict__ out) {
  __shared__ __attribute__((aligned(16))) half_t As[16 * AS_STRIDE];

  const int tid  = threadIdx.x;
  const int wave = tid >> 5;
  const int lane = tid & 31;
  const int hf   = lane >> 4;
  const int l16  = lane & 15;

  const int mt = blockIdx.x >> 3;
  const int nt = (blockIdx.x & 7) * 8 + wave;

  // stage A tile: 16 rows x 1024 halves (2048 x v8h)
#pragma unroll
  for (int it = 0; it < 8; ++it) {
    const int idx = it * 256 + tid;          // 0..2047
    const int row = idx >> 7;                // 0..15
    const int c8  = (idx & 127) << 3;        // 0..1016
    *(v8h*)(&As[row * AS_STRIDE + c8]) =
        *(const v8h*)(A + (size_t)(mt * 16 + row) * E_ + c8);
  }
  __syncthreads();

  const half_t* arow = &As[l16 * AS_STRIDE];
  const float*  wrow = W + (size_t)(nt * 16 + l16) * E_;

  v8f c = {};
  for (int k0 = 0; k0 < E_; k0 += 32) {
    v8h alo = *(const v8h*)(arow + k0 + hf * 16);
    v8h ahi = *(const v8h*)(arow + k0 + hf * 16 + 8);
    v16h a;
#pragma unroll
    for (int e = 0; e < 8; ++e) { a[e] = alo[e]; a[8 + e] = ahi[e]; }
    const float* wp = wrow + k0 + hf * 16;
    v16h bf;
#pragma unroll
    for (int e = 0; e < 16; e += 4) {
      v4f wv = *(const v4f*)(wp + e);
#pragma unroll
      for (int j = 0; j < 4; ++j) bf[e + j] = (half_t)wv[j];
    }
    c = wmma_f16(a, bf, c);
  }
#pragma unroll
  for (int r = 0; r < 8; ++r)
    out[(size_t)(mt * 16 + r + hf * 8) * E_ + nt * 16 + l16] = c[r];
}

extern "C" void kernel_launch(void* const* d_in, const int* in_sizes, int n_in,
                              void* d_out, int out_size, void* d_ws, size_t ws_size,
                              hipStream_t stream) {
  const float* q = (const float*)d_in[0];
  const float* k = (const float*)d_in[1];
  const float* v = (const float*)d_in[2];
  const unsigned char* mask = (const unsigned char*)d_in[3];
  const float* W = (const float*)d_in[4];
  const float* scale = (const float*)d_in[5];
  const float* logC  = (const float*)d_in[6];
  float* out = (float*)d_out;
  half_t* attn = (half_t*)d_ws;   // B*S*E f16 = 16 MB scratch

  attn_kernel<<<B_ * H_ * (S_ / 128), 256, 0, stream>>>(q, k, v, mask, scale,
                                                        logC, attn);
  proj_kernel<<<(B_ * S_ / 16) * 8 / 16, 256, 0, stream>>>(attn, W, out);
}
